// AttentionPooling_29772713296506
// MI455X (gfx1250) — compile-verified, matrix operand order checked by reading, not run on hardware
//
#include <hip/hip_runtime.h>

// Problem constants (from reference)
#define Bsz   16
#define Tlen  8192
#define Ddim  512
#define Kq    4
#define Hh    8
#define HD    64
#define CHUNK 512                 // T-rows per block
#define ROWS  32                  // T-rows per LDS tile / iteration
#define XB_STRIDE 520             // padded bf16 row stride (bank spread)
#define XB_STRIDE32 (XB_STRIDE/2) // row stride in dwords
#define NIT (CHUNK/ROWS)          // 16 iterations per block

typedef __attribute__((ext_vector_type(16))) __bf16   v16bf;
typedef __attribute__((ext_vector_type(8)))  float    v8f;
typedef __attribute__((ext_vector_type(8)))  unsigned v8u;

// 16-bit A/B-operand element mapping (ISA 7.12.2, 16-bit 16x32):
// VGPR v holds a K-pair; lanes 0-15: K in {0..7}U{16..23}, lanes 16-31: +8.
__device__ __forceinline__ int kk_of(int e, int hi) {
    int v = e >> 1, p = e & 1;
    return ((v < 4) ? 0 : 16) + (hi ? 8 : 0) + ((v & 3) << 1) + p;
}
// Same mapping at dword (bf16-pair) granularity.
__device__ __forceinline__ int kp_of(int v, int hi) {
    return ((v < 4) ? 0 : 8) + (hi ? 4 : 0) + (v & 3);
}

__global__ void zero_ws_kernel(float* __restrict__ p, int n) {
    int i = blockIdx.x * 256 + threadIdx.x;
    if (i < n) p[i] = 0.0f;
}

// ---------------------------------------------------------------------------
// Kernel 1: single-pass streaming attention accumulation (register-staged
// double buffer; WMMA bf16 for score + value GEMMs).
//   num[b][k][d] += sum_t exp(q_k.x_t/8) * x[t,d]
//   den[b][h][k] += sum_t exp(q_k.x_t/8)        (masked t contribute 0)
// ---------------------------------------------------------------------------
__global__ __launch_bounds__(256) void attn_stream_kernel(
    const float* __restrict__ x, const int* __restrict__ mask,
    const float* __restrict__ q, float* __restrict__ num, float* __restrict__ den)
{
    __shared__ unsigned xb_u[2][ROWS * XB_STRIDE32];  // x tiles, bf16 packed
    __shared__ unsigned wbuf_u[Hh * 16 * 16];         // exp-weights, A layout

    const int tid   = threadIdx.x;
    const int lane  = tid & 31;
    const int h     = tid >> 5;                 // wave id == head
    const int laneM = lane & 15;
    const int hi    = lane >> 4;

    const int b     = blockIdx.x / (Tlen / CHUNK);
    const int tbase = (blockIdx.x % (Tlen / CHUNK)) * CHUNK;

    // zero weight buffer (rows k=4..15 stay zero forever)
    for (int i = tid; i < Hh * 16 * 16; i += 256) wbuf_u[i] = 0u;

    // Constant A operands for the score GEMM: rows 0..3 = q_k/8, rest 0.
    v16bf qA0, qA1;
    #pragma unroll
    for (int e = 0; e < 16; ++e) {
        int kk = kk_of(e, hi);
        float v0 = 0.0f, v1 = 0.0f;
        if (laneM < Kq) {
            v0 = q[laneM * Ddim + h * HD +      kk] * 0.125f;
            v1 = q[laneM * Ddim + h * HD + 32 + kk] * 0.125f;
        }
        qA0[e] = (__bf16)v0;
        qA1[e] = (__bf16)v1;
    }

    v8f acc[4];                                 // out_h: 4 N-tiles of 16 d's
    #pragma unroll
    for (int j = 0; j < 4; ++j)
        #pragma unroll
        for (int r = 0; r < 8; ++r) acc[j][r] = 0.0f;
    float denAcc[Kq] = {0.0f, 0.0f, 0.0f, 0.0f};

    // ---- register stage: preload tile 0 (16 x global_load_b128 / thread) ----
    const float4* xbase = (const float4*)(x + ((size_t)b * Tlen + tbase) * Ddim);
    float4 stage[16];
    #pragma unroll
    for (int i0 = 0; i0 < 16; ++i0) stage[i0] = xbase[i0 * 256 + tid];

    for (int it = 0; it < NIT; ++it) {
        unsigned* xb  = xb_u[it & 1];
        __bf16*   xbh = (__bf16*)xb;

        // ---- drain stage -> LDS (bf16, merges to ds_store_b64) ----
        #pragma unroll
        for (int i0 = 0; i0 < 16; ++i0) {
            int i   = i0 * 256 + tid;
            int row = i >> 7;                   // 128 float4 per row
            int d   = (i & 127) << 2;
            float4 v = stage[i0];
            __bf16* dst = &xbh[row * XB_STRIDE + d];
            dst[0] = (__bf16)v.x; dst[1] = (__bf16)v.y;
            dst[2] = (__bf16)v.z; dst[3] = (__bf16)v.w;
        }
        __syncthreads();

        // ---- issue next tile's global loads now; they fly during compute ----
        if (it + 1 < NIT) {
            if (it + 2 < NIT) {                 // L2 prefetch one stage further
                const char* p = (const char*)(xbase + (size_t)(it + 2) * (ROWS * Ddim / 4));
                __builtin_prefetch(p + tid * 256, 0, 0);
                __builtin_prefetch(p + tid * 256 + 128, 0, 0);
            }
            const float4* nsrc = xbase + (size_t)(it + 1) * (ROWS * Ddim / 4);
            #pragma unroll
            for (int i0 = 0; i0 < 16; ++i0) stage[i0] = nsrc[i0 * 256 + tid];
        }

        // ---- scores: s(16x16) = qA(16x64) * xb_h^T(64x16), two t-groups ----
        const int t0 = tbase + it * ROWS;
        #pragma unroll
        for (int g = 0; g < 2; ++g) {
            const unsigned* xrow32 = xb + (g * 16 + laneM) * XB_STRIDE32 + (h * HD >> 1);
            v8u u0, u1;
            #pragma unroll
            for (int v = 0; v < 8; ++v) {       // packed b32 LDS reads
                int kp = kp_of(v, hi);
                u0[v] = xrow32[kp];
                u1[v] = xrow32[16 + kp];
            }
            v8f s;
            #pragma unroll
            for (int r = 0; r < 8; ++r) s[r] = 0.0f;
            s = __builtin_amdgcn_wmma_f32_16x16x32_bf16(
                    false, qA0, false, __builtin_bit_cast(v16bf, u0), (short)0, s, false, false);
            s = __builtin_amdgcn_wmma_f32_16x16x32_bf16(
                    false, qA1, false, __builtin_bit_cast(v16bf, u1), (short)0, s, false, false);

            const int t  = t0 + g * 16 + laneM;
            const int mv = mask[b * Tlen + t];
            #pragma unroll
            for (int r = 0; r < Kq; ++r) {      // C: VGPR r, lanes lo -> (k=r, t)
                float wv = (mv != 0) ? __expf(s[r]) : 0.0f;
                if (!hi) ((__bf16*)wbuf_u)[(h * 16 + r) * 32 + g * 16 + laneM] = (__bf16)wv;
                else     wv = 0.0f;             // hi lanes are padding rows
                denAcc[r] += wv;                // reduce across lanes at the end
            }
        }

        // ---- value accumulation: out_h(16x64) += w(16x32) * xb_h(32x64) ----
        const unsigned* wrow32 = wbuf_u + (h * 16 + laneM) * 16;
        v8u ua;
        #pragma unroll
        for (int v = 0; v < 8; ++v) ua[v] = wrow32[kp_of(v, hi)];   // packed b32
        v16bf aw = __builtin_bit_cast(v16bf, ua);
        #pragma unroll
        for (int j = 0; j < 4; ++j) {           // N-tiles of 16 d's
            v16bf bb;                           // B: N = laneM (d), K = t (strided)
            #pragma unroll
            for (int e = 0; e < 16; ++e) {
                int kk = kk_of(e, hi);
                bb[e] = xbh[kk * XB_STRIDE + h * HD + j * 16 + laneM];
            }
            acc[j] = __builtin_amdgcn_wmma_f32_16x16x32_bf16(
                         false, aw, false, bb, (short)0, acc[j], false, false);
        }
    }

    // ---- epilogue: lane reductions + L2 f32 atomics ----
    #pragma unroll
    for (int r = 0; r < Kq; ++r) {
        float wv = denAcc[r];
        for (int off = 16; off > 0; off >>= 1) wv += __shfl_xor(wv, off, 32);
        if (lane == 0) atomicAdd(&den[(b * Hh + h) * Kq + r], wv);
    }
    if (!hi)
        #pragma unroll
        for (int j = 0; j < 4; ++j)
            #pragma unroll
            for (int r = 0; r < Kq; ++r)        // C: VGPR r, lanes lo -> (k=r, d)
                atomicAdd(&num[(b * Kq + r) * Ddim + h * HD + j * 16 + laneM], acc[j][r]);
}

// ---------------------------------------------------------------------------
// Kernel 2: out(64x512) = (num/den)(64x512) @ w_out^T(512x512) + b_out
// One wave per 16x16 output tile, bf16 WMMA over K=512 (16 steps of 32).
// ---------------------------------------------------------------------------
__global__ __launch_bounds__(256) void proj_kernel(
    const float* __restrict__ num, const float* __restrict__ den,
    const float* __restrict__ w_out, const float* __restrict__ b_out,
    float* __restrict__ out)
{
    const int tid   = threadIdx.x;
    const int lane  = tid & 31;
    const int wave  = tid >> 5;
    const int laneM = lane & 15;
    const int hi    = lane >> 4;
    const int tile  = blockIdx.x * 8 + wave;    // 0..127
    const int mtile = tile >> 5;                // 4 row-tiles
    const int ntile = tile & 31;                // 32 col-tiles

    const int mA = mtile * 16 + laneM;          // this lane's A row (m = b*4 + k)
    const int bA = mA >> 2, kA = mA & 3;
    float rden[Hh];
    #pragma unroll
    for (int hh = 0; hh < Hh; ++hh)
        rden[hh] = 1.0f / den[(bA * Hh + hh) * Kq + kA];

    const int nB = ntile * 16 + laneM;          // this lane's B column (n)

    v8f c;
    #pragma unroll
    for (int r = 0; r < 8; ++r) c[r] = 0.0f;

    for (int ds = 0; ds < Ddim / 32; ++ds) {
        const int head = ds >> 1;               // each 32-d step lies in one head
        v16bf a, bm;
        #pragma unroll
        for (int e = 0; e < 16; ++e) {
            int d = ds * 32 + kk_of(e, hi);
            a[e]  = (__bf16)(num[mA * Ddim + d] * rden[head]);   // context row
            bm[e] = (__bf16)w_out[nB * Ddim + d];                // w_out^T column
        }
        c = __builtin_amdgcn_wmma_f32_16x16x32_bf16(false, a, false, bm,
                                                    (short)0, c, false, false);
    }

    const float bias = b_out[ntile * 16 + laneM];
    #pragma unroll
    for (int r = 0; r < 8; ++r) {               // C: VGPR r -> M = r (+8 for hi lanes)
        int m = mtile * 16 + r + (hi ? 8 : 0);
        out[m * Ddim + ntile * 16 + laneM] = c[r] + bias;
    }
}

extern "C" void kernel_launch(void* const* d_in, const int* in_sizes, int n_in,
                              void* d_out, int out_size, void* d_ws, size_t ws_size,
                              hipStream_t stream)
{
    const float* x     = (const float*)d_in[0];
    const int*   mask  = (const int*)  d_in[1];
    const float* q     = (const float*)d_in[2];
    const float* w_out = (const float*)d_in[3];
    const float* b_out = (const float*)d_in[4];

    float* num = (float*)d_ws;                  // [B][K][D]
    float* den = num + Bsz * Kq * Ddim;         // [B][H][K]
    float* out = (float*)d_out;                 // [B][K][D]

    const int nz = Bsz * Kq * Ddim + Bsz * Hh * Kq;
    zero_ws_kernel<<<(nz + 255) / 256, 256, 0, stream>>>(num, nz);
    attn_stream_kernel<<<Bsz * (Tlen / CHUNK), 256, 0, stream>>>(x, mask, q, num, den);
    proj_kernel<<<16, 256, 0, stream>>>(num, den, w_out, b_out, out);
}
